// LIFNet_21483426414798
// MI455X (gfx1250) — compile-verified, matrix operand order checked by reading, not run on hardware
//
#include <hip/hip_runtime.h>
#include <hip/hip_bf16.h>

// ---------------------------------------------------------------------------
// LIF spiking MLP forward on gfx1250 (wave32, WMMA f16 -> f32)
//  - fc1 GEMM computed once (input is T-broadcast over 16 steps)
//  - operands pre-swizzled to fragment-linear layout: each wave-load of a
//    WMMA fragment chunk is 32 lanes x 16B contiguous (512B) from L2
//  - regular (tail-free) software pipeline: chunk kc+1 and kc+2 fragments
//    load under chunk kc / kc+1 wmma; uniform PHIs -> in-place accumulate
//  - LIF1 fused in GEMM epilogue, spikes packed to one uint16 per neuron
//  - fc2 via WMMA with A fragments synthesized from spike bits (exact 0/1)
//  - LIF2 + firing-rate readout fused; tiny MSE-loss reduction kernel
// ---------------------------------------------------------------------------

typedef __attribute__((ext_vector_type(16))) _Float16       v16h;
typedef __attribute__((ext_vector_type(8)))  _Float16       v8h;
typedef __attribute__((ext_vector_type(8)))  float          v8f;
typedef __attribute__((ext_vector_type(4)))  float          v4f;
typedef __attribute__((ext_vector_type(8)))  unsigned short u16x8;

#define D_IN   3072
#define HID    4096
#define OUT_N  10
#define BATCH  512
#define TSTEPS 16
#define KC1    (D_IN / 32)   // 96  K-chunks for fc1
#define KC2    (HID / 32)    // 128 K-chunks for fc2

// Assemble a 16x32-f16 fragment from two swizzled 16B chunks (ISA layout:
// elements 0..7 <- K base, 8..15 <- K base+16, K-half selected by lane>>4).
__device__ __forceinline__ v16h frag_at(const v8h* p) {
  v8h lo = p[0];
  v8h hi = p[32];
  v16h r;
#pragma unroll
  for (int i = 0; i < 8; ++i) { r[i] = lo[i]; r[i + 8] = hi[i]; }
  return r;
}

#define WMMA_F16(A, B, C)                                                     \
  __builtin_amdgcn_wmma_f32_16x16x32_f16(false, (A), false, (B), (short)0,    \
                                         (C), false, false)

#define WMMA_GROUP(A0_, A1_, B0_, B1_, B2_, B3_)                              \
  do {                                                                        \
    acc[0][0] = WMMA_F16(A0_, B0_, acc[0][0]);                                \
    acc[0][1] = WMMA_F16(A0_, B1_, acc[0][1]);                                \
    acc[0][2] = WMMA_F16(A0_, B2_, acc[0][2]);                                \
    acc[0][3] = WMMA_F16(A0_, B3_, acc[0][3]);                                \
    acc[1][0] = WMMA_F16(A1_, B0_, acc[1][0]);                                \
    acc[1][1] = WMMA_F16(A1_, B1_, acc[1][1]);                                \
    acc[1][2] = WMMA_F16(A1_, B2_, acc[1][2]);                                \
    acc[1][3] = WMMA_F16(A1_, B3_, acc[1][3]);                                \
  } while (0)

// ---------------- swizzling converters ---------------------------------
// Fragment-linear layout (v8h units):
//   chunk index = ((tile * KC + kc) * 2 + half) * 32 + lane
// where lane&15 = row/col within tile, lane>>4 = K-half,
//       K base  = kc*32 + half*16 + (lane>>4)*8.

// x [512,3072] f32 -> A-operand swizzle (32 tiles x 96 kc)
__global__ __launch_bounds__(256) void x_swizzle_kernel(
    const float* __restrict__ x, _Float16* __restrict__ xs) {
  const int tid  = blockIdx.x * 256 + threadIdx.x;     // 32*96*2*32 chunks
  const int lane = tid & 31;
  const int half = (tid >> 5) & 1;
  const int kc   = (tid >> 6) % KC1;
  const int mt   = tid / (KC1 * 64);
  const int row  = mt * 16 + (lane & 15);
  const int kb   = kc * 32 + half * 16 + (lane >> 4) * 8;
  const float* src = x + (size_t)row * D_IN + kb;
  v4f s0 = *(const v4f*)(src);
  v4f s1 = *(const v4f*)(src + 4);
  v8h o;
#pragma unroll
  for (int i = 0; i < 4; ++i) { o[i] = (_Float16)s0[i]; o[i + 4] = (_Float16)s1[i]; }
  ((v8h*)xs)[tid] = o;
}

// W1 [3072,4096] f32 -> B-operand swizzle (256 tiles x 96 kc), fused transpose
__global__ __launch_bounds__(256) void w1_swizzle_kernel(
    const float* __restrict__ w1, _Float16* __restrict__ w1s) {
  const int tid  = blockIdx.x * 256 + threadIdx.x;     // 256*96*2*32 chunks
  const int lane = tid & 31;
  const int half = (tid >> 5) & 1;
  const int kc   = (tid >> 6) % KC1;
  const int nt   = tid / (KC1 * 64);
  const int col  = nt * 16 + (lane & 15);
  const int kb   = kc * 32 + half * 16 + (lane >> 4) * 8;
  v8h o;
#pragma unroll
  for (int i = 0; i < 8; ++i)
    o[i] = (_Float16)w1[(size_t)(kb + i) * HID + col];  // 64B coalesced / 16 lanes
  ((v8h*)w1s)[tid] = o;
}

// W2 [4096,10] f32 -> B-operand swizzle (1 tile x 128 kc), cols 10..15 zero
__global__ __launch_bounds__(256) void w2_swizzle_kernel(
    const float* __restrict__ w2, _Float16* __restrict__ w2s) {
  const int tid  = blockIdx.x * 256 + threadIdx.x;     // 128*2*32 chunks
  const int lane = tid & 31;
  const int half = (tid >> 5) & 1;
  const int kc   = tid >> 6;
  const int col  = lane & 15;
  const int kb   = kc * 32 + half * 16 + (lane >> 4) * 8;
  v8h o;
#pragma unroll
  for (int i = 0; i < 8; ++i)
    o[i] = (col < OUT_N) ? (_Float16)w2[(size_t)(kb + i) * OUT_N + col]
                         : (_Float16)0.0f;
  ((v8h*)w2s)[tid] = o;
}

// ---------------- fc1 GEMM + LIF1 + spike bit-pack ---------------------
// Block: 128 threads (4 waves); wave computes 32(M) x 64(N): 2 A frags x
// 4 B frags -> 8 wmma per K-chunk from 12 coalesced 512B wave-loads.
// Grid = (HID/256, BATCH/32) = (16, 16).
__global__ __launch_bounds__(128) void gemm1_lif1_kernel(
    const _Float16* __restrict__ xs,   // swizzled A
    const _Float16* __restrict__ w1s,  // swizzled B
    const float*    __restrict__ b1,   // [4096]
    unsigned short* __restrict__ U) {  // [512,4096] packed spike bits
  const int lane  = threadIdx.x & 31;
  const int wave  = threadIdx.x >> 5;   // 0..3 -> N strip
  const int ln16  = lane & 15;
  const int khalf = lane >> 4;
  const int m0 = blockIdx.y * 32;
  const int n0 = blockIdx.x * 256 + wave * 64;

  const v8h* xs8  = (const v8h*)xs;
  const v8h* w1s8 = (const v8h*)w1s;
  const v8h* ap0 = xs8 + (size_t)((m0 >> 4) + 0) * (KC1 * 64) + lane;
  const v8h* ap1 = xs8 + (size_t)((m0 >> 4) + 1) * (KC1 * 64) + lane;
  const v8h* bp0 = w1s8 + (size_t)((n0 >> 4) + 0) * (KC1 * 64) + lane;
  const v8h* bp1 = w1s8 + (size_t)((n0 >> 4) + 1) * (KC1 * 64) + lane;
  const v8h* bp2 = w1s8 + (size_t)((n0 >> 4) + 2) * (KC1 * 64) + lane;
  const v8h* bp3 = w1s8 + (size_t)((n0 >> 4) + 3) * (KC1 * 64) + lane;

  v8f acc[2][4] = {};

  // Regular, tail-free pipeline: 48 iterations cover 96 chunks; the last
  // prefetch (chunk 96) is a harmless in-workspace over-read.
  v16h ca0 = frag_at(ap0), ca1 = frag_at(ap1);
  v16h cb0 = frag_at(bp0), cb1 = frag_at(bp1);
  v16h cb2 = frag_at(bp2), cb3 = frag_at(bp3);

  for (int kc = 0; kc < KC1; kc += 2) {
    v16h na0 = frag_at(ap0 + 64), na1 = frag_at(ap1 + 64);   // chunk kc+1
    v16h nb0 = frag_at(bp0 + 64), nb1 = frag_at(bp1 + 64);
    v16h nb2 = frag_at(bp2 + 64), nb3 = frag_at(bp3 + 64);
    WMMA_GROUP(ca0, ca1, cb0, cb1, cb2, cb3);                // chunk kc
    ca0 = frag_at(ap0 + 128); ca1 = frag_at(ap1 + 128);      // chunk kc+2
    cb0 = frag_at(bp0 + 128); cb1 = frag_at(bp1 + 128);
    cb2 = frag_at(bp2 + 128); cb3 = frag_at(bp3 + 128);
    WMMA_GROUP(na0, na1, nb0, nb1, nb2, nb3);                // chunk kc+1
    ap0 += 128; ap1 += 128;
    bp0 += 128; bp1 += 128; bp2 += 128; bp3 += 128;
  }

  // Epilogue: bias + 16-step LIF with constant drive, pack spikes to u16.
#pragma unroll
  for (int s = 0; s < 2; ++s) {
#pragma unroll
    for (int j = 0; j < 4; ++j) {
      const int n = n0 + j * 16 + ln16;
      const float bias = b1[n];
#pragma unroll
      for (int r = 0; r < 8; ++r) {
        const float cur = acc[s][j][r] + bias;
        float v = 0.0f;
        unsigned bits = 0u;
#pragma unroll
        for (int t = 0; t < TSTEPS; ++t) {
          v = 0.5f * (v + cur);              // v += (cur - v)/tau, tau = 2
          const bool sp = (v >= 1.0f);       // threshold V_TH = 1
          bits |= sp ? (1u << t) : 0u;
          v = sp ? 0.0f : v;                 // hard reset
        }
        U[(size_t)(m0 + s * 16 + r + 8 * khalf) * HID + n] =
            (unsigned short)bits;
      }
    }
  }
}

// ---------------- fc2 GEMM (spikes x W2) + LIF2 + readout --------------
// Block: 256 threads (8 waves); wave handles one batch b: rows = 16
// timesteps, cols = 16 outputs (10 live). Grid = 512/8 = 64.
__global__ __launch_bounds__(256) void gemm2_lif2_kernel(
    const unsigned short* __restrict__ U,    // [512,4096]
    const _Float16*       __restrict__ w2s,  // swizzled W2^T
    const float*          __restrict__ b2,   // [10]
    float*                __restrict__ pred) {  // [512,10]
  __shared__ float ytile[8][16][17];
  const int lane  = threadIdx.x & 31;
  const int wave  = threadIdx.x >> 5;
  const int ln16  = lane & 15;       // A-row = timestep; D-col = output o
  const int khalf = lane >> 4;
  const int b = blockIdx.x * 8 + wave;

  const unsigned short* ub = U + (size_t)b * HID;
  const v8h* bp = (const v8h*)w2s + lane;

  v8f acc = {};
#pragma unroll 2
  for (int kc = 0; kc < KC2; ++kc) {
    const int klo = kc * 32 + khalf * 8;
    const int khi = kc * 32 + 16 + khalf * 8;
    u16x8 slo = *(const u16x8*)(ub + klo);
    u16x8 shi = *(const u16x8*)(ub + khi);
    v16h a;
#pragma unroll
    for (int i = 0; i < 8; ++i) {
      a[i]     = (_Float16)(float)((slo[i] >> ln16) & 1);
      a[i + 8] = (_Float16)(float)((shi[i] >> ln16) & 1);
    }
    v16h bf = frag_at(bp);
    bp += 64;
    acc = WMMA_F16(a, bf, acc);
  }

  // Stage y[t][o] tile, then LIF2 sequentially over t per output column.
  const float bias = (ln16 < OUT_N) ? b2[ln16] : 0.0f;
#pragma unroll
  for (int r = 0; r < 8; ++r)
    ytile[wave][r + 8 * khalf][ln16] = acc[r] + bias;
  __syncthreads();

  if (lane < 16) {
    const int o = ln16;
    float v = 0.0f, rate = 0.0f;
#pragma unroll
    for (int t = 0; t < TSTEPS; ++t) {
      const float y = ytile[wave][t][o];
      v = 0.5f * (v + y);
      const bool sp = (v >= 1.0f);
      rate += sp ? 1.0f : 0.0f;
      v = sp ? 0.0f : v;
    }
    if (o < OUT_N) pred[(size_t)b * OUT_N + o] = rate * (1.0f / 16.0f);
  }
}

// ---------------- MSE loss reduction -----------------------------------
__global__ __launch_bounds__(256) void loss_kernel(
    const float* __restrict__ pred, const float* __restrict__ lab,
    float* __restrict__ loss) {
  __shared__ float red[256];
  float s = 0.0f;
  for (int i = threadIdx.x; i < BATCH * OUT_N; i += 256) {
    const float d = pred[i] - lab[i];
    s += d * d;
  }
  red[threadIdx.x] = s;
  __syncthreads();
  for (int off = 128; off > 0; off >>= 1) {
    if (threadIdx.x < off) red[threadIdx.x] += red[threadIdx.x + off];
    __syncthreads();
  }
  if (threadIdx.x == 0) *loss = red[0] * (1.0f / (float)(BATCH * OUT_N));
}

// ---------------- launch ------------------------------------------------
extern "C" void kernel_launch(void* const* d_in, const int* in_sizes, int n_in,
                              void* d_out, int out_size, void* d_ws,
                              size_t ws_size, hipStream_t stream) {
  const float* x   = (const float*)d_in[0];  // [512,3,32,32] -> [512,3072]
  const float* lab = (const float*)d_in[1];  // [512,10]
  const float* W1  = (const float*)d_in[2];  // [3072,4096]
  const float* b1  = (const float*)d_in[3];  // [4096]
  const float* W2  = (const float*)d_in[4];  // [4096,10]
  const float* b2  = (const float*)d_in[5];  // [10]
  float* out = (float*)d_out;                // pred(5120) ++ loss(1)

  char* ws = (char*)d_ws;
  _Float16*       xs  = (_Float16*)(ws);                   // 3,145,728 B
  _Float16*       w1s = (_Float16*)(ws + 3145728);         // 25,165,824 B
  _Float16*       w2s = (_Float16*)(ws + 28311552);        // 131,072 B
  unsigned short* U   = (unsigned short*)(ws + 28442624);  // 4,194,304 B

  x_swizzle_kernel<<<(32 * KC1 * 64) / 256, 256, 0, stream>>>(x, xs);
  w1_swizzle_kernel<<<(256 * KC1 * 64) / 256, 256, 0, stream>>>(W1, w1s);
  w2_swizzle_kernel<<<(KC2 * 64) / 256, 256, 0, stream>>>(W2, w2s);
  gemm1_lif1_kernel<<<dim3(HID / 256, BATCH / 32), 128, 0, stream>>>(
      xs, w1s, b1, U);
  gemm2_lif2_kernel<<<BATCH / 8, 256, 0, stream>>>(U, w2s, b2, out);
  loss_kernel<<<1, 256, 0, stream>>>(out, lab, out + BATCH * OUT_N);
}
